// CausalSelfAttention_68126771249479
// MI455X (gfx1250) — compile-verified, matrix operand order checked by reading, not run on hardware
//
#include <hip/hip_runtime.h>
#include <hip/hip_bf16.h>

// ---------------------------------------------------------------- types
typedef __bf16 bf16;
typedef bf16  v16bf __attribute__((ext_vector_type(16)));
typedef float v8f   __attribute__((ext_vector_type(8)));

union FragU { v16bf v; uint4 q[2]; };

__device__ __forceinline__ bf16 f2bf(float f) {
    unsigned u = __builtin_bit_cast(unsigned, f);
    unsigned r = u + 0x7fffu + ((u >> 16) & 1u);   // round-to-nearest-even
    unsigned short h = (unsigned short)(r >> 16);
    return __builtin_bit_cast(bf16, h);
}

// LDS byte offset of a __shared__ object: flat->LDS aperture keeps the offset
// in addr[31:0] (ISA 10.2), and async-to-LDS wants the raw LDS address.
template <typename T>
__device__ __forceinline__ unsigned lds_off(T* p) {
    return (unsigned)(unsigned long long)(p);
}

// GLOBAL_LOAD_ASYNC_TO_LDS_B128: per-lane 16B global -> LDS, tracked by ASYNCcnt.
__device__ __forceinline__ void async_copy16(unsigned lds, const void* g) {
    asm volatile("global_load_async_to_lds_b128 %0, %1, off"
                 :
                 : "v"(lds), "v"(g)
                 : "memory");
}
__device__ __forceinline__ void wait_async0() {
    asm volatile("s_wait_asynccnt 0" ::: "memory");
}

#define Bn 4
#define Tn 2048
#define Cn 1024
#define Hn 16
#define HD 64
#define C3 3072

// ---------------------------------------------------------------- converts
__global__ void k_conv(const float* __restrict__ in, bf16* __restrict__ out, int n) {
    int i = blockIdx.x * blockDim.x + threadIdx.x;
    if (i < n) out[i] = f2bf(in[i]);
}

// in [K][N] f32 (row major) -> out [N][K] bf16 (row major)
__global__ void k_transpose_conv(const float* __restrict__ in, bf16* __restrict__ out,
                                 int K, int N) {
    long i = (long)blockIdx.x * blockDim.x + threadIdx.x;
    long total = (long)K * N;
    if (i < total) {
        int k = (int)(i / N);
        int n = (int)(i % N);
        out[(long)n * K + k] = f2bf(in[i]);
    }
}

// ---------------------------------------------------------------- GEMM (+bias)
// A [M][1024] bf16 row-major, Bt [N][1024] bf16 row-major (i.e. B transposed),
// out = A * Bt^T + bias : [M][N].  Block tile 128x128, BK=32, 256 threads (8 waves).
// Double-buffered LDS staged via GLOBAL_LOAD_ASYNC_TO_LDS_B128 (ASYNCcnt).
template <bool OUT_BF16>
__global__ __launch_bounds__(256) void k_gemm_bias(const bf16* __restrict__ A,
                                                   const bf16* __restrict__ Bt,
                                                   const float* __restrict__ bias,
                                                   void* __restrict__ outp, int N) {
    constexpr int K = 1024, BK = 32, NK = K / BK;
    __shared__ __align__(16) bf16 As[2][128][40];
    __shared__ __align__(16) bf16 Bs[2][128][40];

    const int tid  = threadIdx.x;
    const int lane = tid & 31, wave = tid >> 5;
    const int half = lane >> 4, l16 = lane & 15;
    const int waveM = wave >> 2, waveN = wave & 3;          // 2 x 4 wave grid
    const long m0 = (long)blockIdx.y * 128;
    const long n0 = (long)blockIdx.x * 128;

    // this thread's two 16B chunks within a 128x32 tile
    const int r0 = tid >> 2,          c80 = (tid & 3) * 8;          // chunk tid
    const int r1 = (tid + 256) >> 2,  c81 = ((tid + 256) & 3) * 8;  // chunk tid+256

    v8f acc[4][2];
#pragma unroll
    for (int i = 0; i < 4; i++)
#pragma unroll
        for (int j = 0; j < 2; j++) acc[i][j] = (v8f)(0.0f);

    auto issue = [&](int buf, int k0) {
        async_copy16(lds_off(&As[buf][r0][c80]), &A[(m0 + r0) * K + k0 + c80]);
        async_copy16(lds_off(&Bs[buf][r0][c80]), &Bt[(n0 + r0) * K + k0 + c80]);
        async_copy16(lds_off(&As[buf][r1][c81]), &A[(m0 + r1) * K + k0 + c81]);
        async_copy16(lds_off(&Bs[buf][r1][c81]), &Bt[(n0 + r1) * K + k0 + c81]);
    };

    issue(0, 0);
    for (int i = 0; i < NK; i++) {
        const int cur = i & 1;
        wait_async0();          // my async writes into buf[cur] have landed
        __syncthreads();        // everyone's landed; everyone done reading buf[cur^1]
        if (i + 1 < NK) issue(cur ^ 1, (i + 1) * BK);   // overlaps with compute below

        FragU a[4], b[2];
#pragma unroll
        for (int mt = 0; mt < 4; mt++) {
            int row = waveM * 64 + mt * 16 + l16;
            a[mt].q[0] = *(const uint4*)&As[cur][row][half * 8];
            a[mt].q[1] = *(const uint4*)&As[cur][row][16 + half * 8];
        }
#pragma unroll
        for (int nt = 0; nt < 2; nt++) {
            int row = waveN * 32 + nt * 16 + l16;
            b[nt].q[0] = *(const uint4*)&Bs[cur][row][half * 8];
            b[nt].q[1] = *(const uint4*)&Bs[cur][row][16 + half * 8];
        }
#pragma unroll
        for (int mt = 0; mt < 4; mt++)
#pragma unroll
            for (int nt = 0; nt < 2; nt++)
                acc[mt][nt] = __builtin_amdgcn_wmma_f32_16x16x32_bf16(
                    false, a[mt].v, false, b[nt].v, (short)0, acc[mt][nt], false, false);
    }

    // epilogue: C layout row = r + 8*half, col = l16
#pragma unroll
    for (int mt = 0; mt < 4; mt++) {
#pragma unroll
        for (int nt = 0; nt < 2; nt++) {
            long col = n0 + waveN * 32 + nt * 16 + l16;
            float bv = bias[col];
#pragma unroll
            for (int r = 0; r < 8; r++) {
                long row = m0 + waveM * 64 + mt * 16 + r + half * 8;
                float v  = acc[mt][nt][r] + bv;
                if constexpr (OUT_BF16)
                    ((bf16*)outp)[row * (long)N + col] = f2bf(v);
                else
                    ((float*)outp)[row * (long)N + col] = v;
            }
        }
    }
}

// ---------------------------------------------------------------- flash attention
// qkv [B*T][3072] bf16 (Q|K|V per row, H heads x 64 dims).  One block = 128 query
// rows of one (b,h); 8 waves, each wave owns 16 rows.  Online softmax, causal.
__global__ __launch_bounds__(256) void k_attn(const bf16* __restrict__ qkv,
                                              bf16* __restrict__ y) {
    __shared__ __align__(16) bf16 Vt[64][136];      // V transposed: [dim][key]
    __shared__ __align__(16) bf16 Ps[8][16][136];   // per-wave P tile [row][key]

    const int tid  = threadIdx.x;
    const int lane = tid & 31, wave = tid >> 5;
    const int half = lane >> 4, l16 = lane & 15;
    const int qb = blockIdx.x, h = blockIdx.y, b = blockIdx.z;

    // Q A-fragments: row = qb*128 + wave*16 + l16, contraction = head dim (64 = 2 steps)
    const long rowQ  = (long)qb * 128 + wave * 16 + l16;
    const long baseQ = ((long)b * Tn + rowQ) * C3 + (long)h * HD;
    FragU aQ[2];
#pragma unroll
    for (int kk = 0; kk < 2; kk++) {
        aQ[kk].q[0] = *(const uint4*)&qkv[baseQ + kk * 32 + half * 8];
        aQ[kk].q[1] = *(const uint4*)&qkv[baseQ + kk * 32 + 16 + half * 8];
    }

    float m_i[8], l_i[8];
    v8f o[4];
#pragma unroll
    for (int r = 0; r < 8; r++) { m_i[r] = -1e30f; l_i[r] = 0.0f; }
#pragma unroll
    for (int d = 0; d < 4; d++) o[d] = (v8f)(0.0f);

    for (int kb = 0; kb <= qb; kb++) {
        // ---- stage V tile transposed into LDS: thread t -> key t/2, dims (t&1)*32..+31
        {
            int k  = tid >> 1;
            int dh = (tid & 1) * 32;
            const bf16* vp = &qkv[((long)b * Tn + (long)kb * 128 + k) * C3 + 2 * Cn +
                                  (long)h * HD + dh];
            union { uint4 q[4]; bf16 e[32]; } tmp;
#pragma unroll
            for (int i = 0; i < 4; i++) tmp.q[i] = *(const uint4*)(vp + i * 8);
#pragma unroll
            for (int d = 0; d < 32; d++) Vt[dh + d][k] = tmp.e[d];
        }
        __syncthreads();

        // ---- prefetch next key block's K and V rows into L2/L0 (global_prefetch_b8)
        if (kb + 1 <= qb) {
            long nrow = (long)b * Tn + (long)(kb + 1) * 128 + (tid >> 1);
            const bf16* pk = &qkv[nrow * C3 + Cn + (long)h * HD + (tid & 1) * 32];
            const bf16* pv = &qkv[nrow * C3 + 2 * Cn + (long)h * HD + (tid & 1) * 32];
            __builtin_prefetch(pk, 0, 0);
            __builtin_prefetch(pv, 0, 0);
        }

        // ---- S = Q * K^T  (16 rows x 128 keys per wave); K B-frags direct from global
        v8f s[8];
#pragma unroll
        for (int nt = 0; nt < 8; nt++) {
            long keyrow = (long)kb * 128 + nt * 16 + l16;
            const bf16* kp = &qkv[((long)b * Tn + keyrow) * C3 + Cn + (long)h * HD];
            FragU bk[2];
#pragma unroll
            for (int kk = 0; kk < 2; kk++) {
                bk[kk].q[0] = *(const uint4*)(kp + kk * 32 + half * 8);
                bk[kk].q[1] = *(const uint4*)(kp + kk * 32 + 16 + half * 8);
            }
            v8f t = (v8f)(0.0f);
            t = __builtin_amdgcn_wmma_f32_16x16x32_bf16(false, aQ[0].v, false, bk[0].v,
                                                        (short)0, t, false, false);
            t = __builtin_amdgcn_wmma_f32_16x16x32_bf16(false, aQ[1].v, false, bk[1].v,
                                                        (short)0, t, false, false);
            s[nt] = t;
        }

        // ---- causal mask + scale + row max (reduce across the 16 lanes of this half)
        float rmax[8];
#pragma unroll
        for (int r = 0; r < 8; r++) {
            long row = (long)qb * 128 + wave * 16 + r + half * 8;
            float mx = -1e30f;
#pragma unroll
            for (int nt = 0; nt < 8; nt++) {
                long col = (long)kb * 128 + nt * 16 + l16;
                float v  = s[nt][r] * 0.125f;          // 1/sqrt(64)
                v        = (col <= row) ? v : -1e30f;
                s[nt][r] = v;
                mx       = fmaxf(mx, v);
            }
#pragma unroll
            for (int m = 1; m < 16; m <<= 1) mx = fmaxf(mx, __shfl_xor(mx, m, 32));
            rmax[r] = mx;
        }

        // ---- online softmax update
#pragma unroll
        for (int r = 0; r < 8; r++) {
            float m_new = fmaxf(m_i[r], rmax[r]);
            float alpha = __expf(m_i[r] - m_new);
            float rsum  = 0.0f;
#pragma unroll
            for (int nt = 0; nt < 8; nt++) {
                float p  = __expf(s[nt][r] - m_new);
                s[nt][r] = p;
                rsum += p;
            }
#pragma unroll
            for (int m = 1; m < 16; m <<= 1) rsum += __shfl_xor(rsum, m, 32);
            l_i[r] = l_i[r] * alpha + rsum;
            m_i[r] = m_new;
#pragma unroll
            for (int d = 0; d < 4; d++) o[d][r] *= alpha;
        }

        // ---- P -> LDS (C-layout -> A-layout relayout through LDS)
#pragma unroll
        for (int nt = 0; nt < 8; nt++)
#pragma unroll
            for (int r = 0; r < 8; r++)
                Ps[wave][r + half * 8][nt * 16 + l16] = f2bf(s[nt][r]);
        __syncthreads();

        // ---- O += P * V  (contraction over 128 keys = 4 steps)
        FragU aP[4];
#pragma unroll
        for (int kk = 0; kk < 4; kk++) {
            aP[kk].q[0] = *(const uint4*)&Ps[wave][l16][kk * 32 + half * 8];
            aP[kk].q[1] = *(const uint4*)&Ps[wave][l16][kk * 32 + 16 + half * 8];
        }
#pragma unroll
        for (int dt = 0; dt < 4; dt++) {
            v8f t = o[dt];
#pragma unroll
            for (int kk = 0; kk < 4; kk++) {
                FragU bv;
                bv.q[0] = *(const uint4*)&Vt[dt * 16 + l16][kk * 32 + half * 8];
                bv.q[1] = *(const uint4*)&Vt[dt * 16 + l16][kk * 32 + 16 + half * 8];
                t = __builtin_amdgcn_wmma_f32_16x16x32_bf16(false, aP[kk].v, false, bv.v,
                                                            (short)0, t, false, false);
            }
            o[dt] = t;
        }
        __syncthreads();   // protect Vt/Ps before next key block
    }

    // ---- normalize + store bf16 y[b*T+row][h*64+d]
#pragma unroll
    for (int r = 0; r < 8; r++) {
        float inv = 1.0f / l_i[r];
        long row  = (long)qb * 128 + wave * 16 + r + half * 8;
        bf16* yp  = &y[((long)b * Tn + row) * Cn + (long)h * HD];
#pragma unroll
        for (int dt = 0; dt < 4; dt++)
            yp[dt * 16 + l16] = f2bf(o[dt][r] * inv);
    }
}

// ---------------------------------------------------------------- launch
extern "C" void kernel_launch(void* const* d_in, const int* in_sizes, int n_in,
                              void* d_out, int out_size, void* d_ws, size_t ws_size,
                              hipStream_t stream) {
    const float* x      = (const float*)d_in[0];
    const float* w_attn = (const float*)d_in[1];
    const float* b_attn = (const float*)d_in[2];
    const float* w_proj = (const float*)d_in[3];
    const float* b_proj = (const float*)d_in[4];
    float* out = (float*)d_out;

    bf16* xb  = (bf16*)d_ws;                       // 8192*1024
    bf16* waT = xb  + (size_t)8192 * 1024;         // 3072*1024 (w_attn^T)
    bf16* wpT = waT + (size_t)3072 * 1024;         // 1024*1024 (w_proj^T)
    bf16* qkv = wpT + (size_t)1024 * 1024;         // 8192*3072
    bf16* yb  = qkv + (size_t)8192 * 3072;         // 8192*1024

    {   // x -> bf16
        int n = 8192 * 1024;
        k_conv<<<(n + 255) / 256, 256, 0, stream>>>(x, xb, n);
    }
    {   // w_attn [1024][3072] -> waT [3072][1024] bf16
        long n = 1024L * 3072;
        k_transpose_conv<<<(unsigned)((n + 255) / 256), 256, 0, stream>>>(w_attn, waT, 1024, 3072);
    }
    {   // w_proj [1024][1024] -> wpT [1024][1024] bf16
        long n = 1024L * 1024;
        k_transpose_conv<<<(unsigned)((n + 255) / 256), 256, 0, stream>>>(w_proj, wpT, 1024, 1024);
    }

    // QKV projection: [8192x1024] * [1024x3072] + b_attn -> qkv (bf16)
    k_gemm_bias<true><<<dim3(3072 / 128, 8192 / 128), 256, 0, stream>>>(
        xb, waT, b_attn, (void*)qkv, 3072);

    // flash attention -> yb (bf16)
    k_attn<<<dim3(Tn / 128, Hn, Bn), 256, 0, stream>>>(qkv, yb);

    // output projection: [8192x1024] * [1024x1024] + b_proj -> out (f32)
    k_gemm_bias<false><<<dim3(1024 / 128, 8192 / 128), 256, 0, stream>>>(
        yb, wpT, b_proj, (void*)out, 1024);
}
